// CustomDeepseekV2MoE_32495722561864
// MI455X (gfx1250) — matrix operand order repaired and checked by reading, not compiled, not run on hardware
//
#include <hip/hip_runtime.h>
#include <hip/hip_bf16.h>

#define T_TOK 1024
#define DDIM  2048
#define NEXP  32
#define IDIM  1408
#define KSEL  6
#define NGRP  8
#define TOPG  3
#define SIDIM 2816
#define RSCALE 2.5f

typedef __attribute__((ext_vector_type(16))) __bf16 v16bf;
typedef __attribute__((ext_vector_type(8)))  float  v8f;
typedef __attribute__((ext_vector_type(2)))  float  f32x2;
typedef __attribute__((ext_vector_type(2)))  __bf16 bf16x2;
typedef __attribute__((ext_vector_type(4)))  unsigned u32x4;

static __device__ __forceinline__ float sig_(float v) { return 1.0f / (1.0f + __expf(-v)); }
static __device__ __forceinline__ float sig_fast_(float v) {
    return __builtin_amdgcn_rcpf(1.0f + __expf(-v));
}

static __device__ __forceinline__ unsigned pack_bf16x2(float a, float b) {
    f32x2 v = {a, b};
    bf16x2 r = __builtin_convertvector(v, bf16x2);   // v_cvt_pk_bf16_f32
    return __builtin_bit_cast(unsigned, r);
}

static __device__ __forceinline__ v8f zero8() {
    v8f z = {0.f, 0.f, 0.f, 0.f, 0.f, 0.f, 0.f, 0.f};
    return z;
}

// Async global->LDS 128-bit copy (CDNA5, tracked by ASYNCcnt).  lds_ptr must
// be an LDS (shared) pointer; we pass its 32-bit LDS byte offset in a VGPR.
static __device__ __forceinline__ void async_load_b128(void* lds_ptr, const void* gptr) {
    unsigned loff = (unsigned)(uintptr_t)(__attribute__((address_space(3))) char*)lds_ptr;
    asm volatile("global_load_async_to_lds_b128 %0, %1, off"
                 :: "v"(loff), "v"(gptr) : "memory");
}
static __device__ __forceinline__ void wait_async0() {
    asm volatile("s_wait_asynccnt 0x0" ::: "memory");
}

// ---------------------------------------------------------------------------
// Routing: 1 wave (32 lanes) per token; lane e computes expert e's logit.
// ---------------------------------------------------------------------------
__global__ void route_kernel(const float* __restrict__ x,
                             const float* __restrict__ gw,
                             const float* __restrict__ bias,
                             int* __restrict__ cnt,
                             int* __restrict__ tok_list,
                             float* __restrict__ w_list) {
    int t = blockIdx.x;
    int e = threadIdx.x;  // 0..31
    const float* xr = x + (size_t)t * DDIM;
    float acc = 0.f;
    for (int d = 0; d < DDIM; ++d)
        acc = fmaf(xr[d], gw[(size_t)d * NEXP + e], acc);
    float sc = sig_(acc);
    float co = sc + bias[e];
    __shared__ float s_sc[NEXP], s_co[NEXP];
    s_sc[e] = sc; s_co[e] = co;
    __syncthreads();
    if (e == 0) {
        float gs[NGRP];
        for (int g = 0; g < NGRP; ++g) {
            float a[4];
            for (int i = 0; i < 4; ++i) a[i] = s_co[g * 4 + i];
            int im = 0;
            for (int i = 1; i < 4; ++i) if (a[i] > a[im]) im = i;
            float m1 = a[im]; a[im] = -INFINITY;
            int im2 = 0;
            for (int i = 1; i < 4; ++i) if (a[i] > a[im2]) im2 = i;
            gs[g] = m1 + a[im2];
        }
        bool gsel[NGRP];
        for (int g = 0; g < NGRP; ++g) gsel[g] = false;
        for (int s = 0; s < TOPG; ++s) {
            int bi = -1; float bv = -INFINITY;
            for (int g = 0; g < NGRP; ++g)
                if (!gsel[g] && gs[g] > bv) { bv = gs[g]; bi = g; }
            gsel[bi] = true;
        }
        bool used[NEXP];
        for (int i = 0; i < NEXP; ++i) used[i] = false;
        int sel[KSEL]; float selw[KSEL]; float wsum = 0.f;
        for (int s = 0; s < KSEL; ++s) {
            int bi = -1; float bv = -INFINITY;
            for (int ee = 0; ee < NEXP; ++ee) {
                if (!gsel[ee >> 2] || used[ee]) continue;
                if (s_co[ee] > bv) { bv = s_co[ee]; bi = ee; }
            }
            used[bi] = true; sel[s] = bi; selw[s] = s_sc[bi]; wsum += s_sc[bi];
        }
        float inv = RSCALE / (wsum + 1e-20f);
        for (int s = 0; s < KSEL; ++s) {
            int ee = sel[s];
            int i = atomicAdd(&cnt[ee], 1);
            tok_list[ee * T_TOK + i] = t;
            w_list[ee * T_TOK + i]   = selw[s] * inv;
        }
    }
}

__global__ void scan_kernel(const int* __restrict__ cnt, int* __restrict__ offs) {
    if (threadIdx.x == 0) {
        int s = 0;
        for (int e = 0; e < NEXP; ++e) { offs[e] = s; s += cnt[e]; }
    }
}

// ---------------------------------------------------------------------------
// A-fragment loader: row-major LDS tile, row stride RSTR words (padded +4 so
// row r starts at bank 4r -> two conflict-free ds_load_b128 per fragment).
// ---------------------------------------------------------------------------
template <int RSTR>
static __device__ __forceinline__ v16bf load_afrag(const unsigned* __restrict__ A32,
                                                   int row, int kword_base) {
    union { u32x4 q[2]; v16bf v; } af;
    const u32x4* p = (const u32x4*)(A32 + row * RSTR + kword_base);
    af.q[0] = p[0];   // words base+0..3
    af.q[1] = p[1];   // words base+4..7
    return af.v;
}

// ---------------------------------------------------------------------------
// Fused gate+up GEMM + SiLU-mul.  Block = 256 threads (8 waves, wave32).
// ---------------------------------------------------------------------------
template <bool ROUTED, int LDN>
__global__ __launch_bounds__(256, 1)
void gateup_kernel(const float* __restrict__ x,
                   const float* __restrict__ wg,
                   const float* __restrict__ wu,
                   const int* __restrict__ cnt,
                   const int* __restrict__ offs,
                   const int* __restrict__ tok_list,
                   __bf16* __restrict__ H) {
    constexpr int KTOT = DDIM;
    constexpr int KW   = KTOT / 2;      // 1024 words per row
    constexpr int RSTR = KW + 4;        // padded row stride (words)
    extern __shared__ char smem[];
    unsigned* A32 = (unsigned*)smem;                    // 64*RSTR words
    int*      t_s = (int*)(smem + (size_t)64 * RSTR * 4);
    int tid = threadIdx.x, lane = tid & 31, wv = tid >> 5;

    int ne, rbase, e = 0;
    if constexpr (ROUTED) {
        e = blockIdx.x;
        ne = cnt[e];
        int row0 = blockIdx.y * 64;
        if (row0 >= ne) return;
        if (tid < 64) {
            int pr = row0 + tid;
            int prc = pr < ne ? pr : ne - 1;
            t_s[tid] = tok_list[e * T_TOK + prc];
        }
        rbase = offs[e] + row0;
        ne -= row0;
    } else {
        int row0 = blockIdx.x * 64;
        if (tid < 64) t_s[tid] = row0 + tid;
        rbase = row0; ne = 64;
    }
    __syncthreads();

    // Stage A (fp32 -> packed bf16)
    for (int idx = tid; idx < 64 * KW; idx += 256) {
        int r = idx / KW, wi = idx - r * KW;
        const f32x2 v = *(const f32x2*)(x + (size_t)t_s[r] * DDIM + wi * 2);
        A32[r * RSTR + wi] = pack_bf16x2(v.x, v.y);
    }
    __syncthreads();

    const size_t wbase = ROUTED ? (size_t)e * KTOT * LDN : 0;
    int nb    = (blockIdx.z * 8 + wv) * 16;
    int ncol  = lane & 15;
    int khalf = lane >> 4;  // 0 or 1
    const float* pg = wg + wbase + (size_t)(khalf * 16) * LDN + nb + ncol;
    const float* pu = wu + wbase + (size_t)(khalf * 16) * LDN + nb + ncol;

    v8f accg[4], accu[4];
    for (int s = 0; s < 4; ++s) { accg[s] = zero8(); accu[s] = zero8(); }

    for (int k0 = 0; k0 < KTOT; k0 += 32) {
        union { unsigned u[8]; v16bf v; } bg, bu;
#pragma unroll
        for (int j = 0; j < 8; ++j) {
            bg.u[j] = pack_bf16x2(pg[(2 * j) * LDN], pg[(2 * j + 1) * LDN]);
            bu.u[j] = pack_bf16x2(pu[(2 * j) * LDN], pu[(2 * j + 1) * LDN]);
        }
        if (k0 + 64 < KTOT) {  // pull next-next rows of the weight stream
            __builtin_prefetch(pg + (size_t)64 * LDN, 0, 0);
            __builtin_prefetch(pu + (size_t)64 * LDN, 0, 0);
        }
        int kwb = (k0 >> 1) + khalf * 8;
        // Preload all A fragments, then burst the 8 WMMAs back-to-back so the
        // scheduler can use partial dscnt waits and XDL co-issue.
        v16bf a[4];
#pragma unroll
        for (int s = 0; s < 4; ++s)
            a[s] = load_afrag<RSTR>(A32, s * 16 + (lane & 15), kwb);
#pragma unroll
        for (int s = 0; s < 4; ++s) {
            accg[s] = __builtin_amdgcn_wmma_f32_16x16x32_bf16(false, a[s], false, bg.v,
                                                              (short)0, accg[s], false, false);
            accu[s] = __builtin_amdgcn_wmma_f32_16x16x32_bf16(false, a[s], false, bu.v,
                                                              (short)0, accu[s], false, false);
        }
        pg += (size_t)32 * LDN;
        pu += (size_t)32 * LDN;
    }

    // Epilogue: H = silu(gate) * up  (C layout: VGPR r -> M=r (lanes<16) / 8+r)
    for (int s = 0; s < 4; ++s) {
#pragma unroll
        for (int r2 = 0; r2 < 8; ++r2) {
            int M = (lane < 16) ? r2 : 8 + r2;
            int row = s * 16 + M;
            if (row < ne) {
                float g = accg[s][r2], u = accu[s][r2];
                float h = g * sig_fast_(g) * u;
                H[(size_t)(rbase + row) * LDN + nb + ncol] = (__bf16)h;
            }
        }
    }
}

// ---------------------------------------------------------------------------
// Down projection: out[t,:] (+)= w * (H_row @ w_down).  K staged in CK chunks
// via async global->LDS copies (H is already bf16; raw byte copy).
// ---------------------------------------------------------------------------
template <bool ROUTED, int KTOT, int CK>
__global__ __launch_bounds__(256, 1)
void down_kernel(const __bf16* __restrict__ H,
                 const float* __restrict__ wd,
                 const int* __restrict__ cnt,
                 const int* __restrict__ offs,
                 const int* __restrict__ tok_list,
                 const float* __restrict__ w_list,
                 float* __restrict__ out) {
    constexpr int CKW  = CK / 2;       // 32-bit words per row
    constexpr int CQ   = CKW / 4;      // 128-bit chunks per row
    constexpr int RSTR = CKW + 4;      // padded row stride (words)
    extern __shared__ char smem[];
    unsigned* A32 = (unsigned*)smem;   // 64*RSTR words
    int*      t_s = (int*)(smem + (size_t)64 * RSTR * 4);
    float*    w_s = (float*)(t_s + 64);
    int tid = threadIdx.x, lane = tid & 31, wv = tid >> 5;

    int ne, hbase, e = 0;
    if constexpr (ROUTED) {
        e = blockIdx.x;
        ne = cnt[e];
        int row0 = blockIdx.y * 64;
        if (row0 >= ne) return;
        if (tid < 64) {
            int pr = row0 + tid;
            int prc = pr < ne ? pr : ne - 1;
            t_s[tid] = tok_list[e * T_TOK + prc];
            w_s[tid] = (pr < ne) ? w_list[e * T_TOK + pr] : 0.f;
        }
        hbase = offs[e] + row0;
        ne -= row0;
    } else {
        int row0 = blockIdx.x * 64;
        if (tid < 64) { t_s[tid] = row0 + tid; w_s[tid] = 1.f; }
        hbase = row0; ne = 64;
    }
    __syncthreads();

    const float* WD = ROUTED ? wd + (size_t)e * KTOT * DDIM : wd;
    int nb    = (blockIdx.z * 8 + wv) * 16;
    int ncol  = lane & 15;
    int khalf = lane >> 4;
    v8f acc[4];
    for (int s = 0; s < 4; ++s) acc[s] = zero8();

    for (int c0 = 0; c0 < KTOT; c0 += CK) {
        if (c0) __syncthreads();
        // Async-stage the 64 x CK bf16 H tile (rows contiguous per segment).
        for (int idx = tid; idx < 64 * CQ; idx += 256) {
            int r = idx / CQ, q = idx - r * CQ;
            int rc = r < ne ? r : ne - 1;
            const __bf16* g = H + (size_t)(hbase + rc) * KTOT + c0 + q * 8;
            async_load_b128(A32 + r * RSTR + q * 4, g);
        }
        wait_async0();
        __syncthreads();
        const float* pb = WD + (size_t)(c0 + khalf * 16) * DDIM + nb + ncol;
        for (int kk = 0; kk < CK; kk += 32) {
            union { unsigned u[8]; v16bf v; } bfr;
#pragma unroll
            for (int j = 0; j < 8; ++j)
                bfr.u[j] = pack_bf16x2(pb[(2 * j) * DDIM], pb[(2 * j + 1) * DDIM]);
            if (kk + 64 < CK) __builtin_prefetch(pb + (size_t)64 * DDIM, 0, 0);
            int kwb = (kk >> 1) + khalf * 8;
            v16bf a[4];
#pragma unroll
            for (int s = 0; s < 4; ++s)
                a[s] = load_afrag<RSTR>(A32, s * 16 + (lane & 15), kwb);
#pragma unroll
            for (int s = 0; s < 4; ++s)
                acc[s] = __builtin_amdgcn_wmma_f32_16x16x32_bf16(false, a[s], false, bfr.v,
                                                                 (short)0, acc[s], false, false);
            pb += (size_t)32 * DDIM;
        }
    }

    for (int s = 0; s < 4; ++s) {
#pragma unroll
        for (int r2 = 0; r2 < 8; ++r2) {
            int M = (lane < 16) ? r2 : 8 + r2;
            int row = s * 16 + M;
            if (row < ne) {
                float y = acc[s][r2];
                int t = t_s[row];
                if constexpr (ROUTED)
                    atomicAdd(&out[(size_t)t * DDIM + nb + ncol], y * w_s[row]);
                else
                    out[(size_t)t * DDIM + nb + ncol] = y;
            }
        }
    }
}

// ---------------------------------------------------------------------------
extern "C" void kernel_launch(void* const* d_in, const int* in_sizes, int n_in,
                              void* d_out, int out_size, void* d_ws, size_t ws_size,
                              hipStream_t stream) {
    const float* x       = (const float*)d_in[0];
    const float* gate_w  = (const float*)d_in[1];
    const float* ebias   = (const float*)d_in[2];
    const float* w_gate  = (const float*)d_in[3];
    const float* w_up    = (const float*)d_in[4];
    const float* w_down  = (const float*)d_in[5];
    const float* ws_gate = (const float*)d_in[6];
    const float* ws_up   = (const float*)d_in[7];
    const float* ws_down = (const float*)d_in[8];
    float* out = (float*)d_out;

    // Workspace layout
    char* ws = (char*)d_ws;
    int*    cnt      = (int*)ws;                       ws += 256;
    int*    offs     = (int*)ws;                       ws += 256;
    int*    tok_list = (int*)ws;                       ws += (size_t)NEXP * T_TOK * 4;
    float*  w_list   = (float*)ws;                     ws += (size_t)NEXP * T_TOK * 4;
    __bf16* Hr       = (__bf16*)ws;                    ws += (size_t)T_TOK * KSEL * IDIM * 2;
    __bf16* Hs       = (__bf16*)ws;

    const int smem_gu   = 64 * (DDIM / 2 + 4) * 4 + 64 * 4;            // 263424 B
    const int smem_down = 64 * (IDIM / 2 + 4) * 4 + 64 * 4 + 64 * 4;   // 181760 B
    hipFuncSetAttribute((const void*)gateup_kernel<true, IDIM>,
                        hipFuncAttributeMaxDynamicSharedMemorySize, smem_gu);
    hipFuncSetAttribute((const void*)gateup_kernel<false, SIDIM>,
                        hipFuncAttributeMaxDynamicSharedMemorySize, smem_gu);
    hipFuncSetAttribute((const void*)down_kernel<true, IDIM, IDIM>,
                        hipFuncAttributeMaxDynamicSharedMemorySize, smem_down);
    hipFuncSetAttribute((const void*)down_kernel<false, SIDIM, IDIM>,
                        hipFuncAttributeMaxDynamicSharedMemorySize, smem_down);

    hipMemsetAsync(cnt, 0, 256, stream);

    route_kernel<<<T_TOK, 32, 0, stream>>>(x, gate_w, ebias, cnt, tok_list, w_list);
    scan_kernel<<<1, 32, 0, stream>>>(cnt, offs);

    // Shared expert gate+up:  [1024,2048] @ [2048,2816] -> Hs (bf16)
    gateup_kernel<false, SIDIM><<<dim3(T_TOK / 64, 1, SIDIM / 128), 256, smem_gu, stream>>>(
        x, ws_gate, ws_up, nullptr, nullptr, nullptr, Hs);
    // Routed experts gate+up (gathered tokens per expert) -> Hr (bf16)
    gateup_kernel<true, IDIM><<<dim3(NEXP, T_TOK / 64, IDIM / 128), 256, smem_gu, stream>>>(
        x, w_gate, w_up, cnt, offs, tok_list, Hr);

    // Shared down-proj writes out directly (full coverage)
    down_kernel<false, SIDIM, IDIM><<<dim3(T_TOK / 64, 1, DDIM / 128), 256, smem_down, stream>>>(
        Hs, ws_down, nullptr, nullptr, nullptr, nullptr, out);
    // Routed down-proj: scaled by combine weight, atomically accumulated
    down_kernel<true, IDIM, IDIM><<<dim3(NEXP, T_TOK / 64, DDIM / 128), 256, smem_down, stream>>>(
        Hr, w_down, cnt, offs, tok_list, w_list, out);
}
